// FP8Linear_20444044329513
// MI455X (gfx1250) — 8-bit matrix operand order repaired and checked by reading, compiled, not run on hardware
//
#include <hip/hip_runtime.h>
#include <hip/hip_bf16.h>

typedef __attribute__((ext_vector_type(16))) int   v16i;
typedef __attribute__((ext_vector_type(8)))  float v8f;
typedef int v4i_gcc __attribute__((vector_size(16)));

#define FP8_EPS 1e-6f
#define AS1 __attribute__((address_space(1)))
#define AS3 __attribute__((address_space(3)))

#if __has_builtin(__builtin_amdgcn_global_load_async_to_lds_b128) && \
    __has_builtin(__builtin_amdgcn_s_wait_asynccnt)
#define USE_ASYNC_LDS 1
#else
#define USE_ASYNC_LDS 0
#endif

__global__ void fp8lin_init(unsigned* hdr) {
    if (threadIdx.x < 2) hdr[threadIdx.x] = 0u;
}

__global__ void fp8lin_amax(const unsigned* __restrict__ in_u32, long n_elem,
                            unsigned* __restrict__ slot) {
    __shared__ float red[256];
    float m = 0.f;
    long n32 = n_elem >> 1;
    long i = (long)blockIdx.x * blockDim.x + threadIdx.x;
    long stride = (long)gridDim.x * blockDim.x;
    for (; i < n32; i += stride) {
        unsigned u = in_u32[i];
        float f0 = __uint_as_float(u << 16);
        float f1 = __uint_as_float(u & 0xFFFF0000u);
        m = fmaxf(m, fmaxf(fabsf(f0), fabsf(f1)));
    }
    if (blockIdx.x == 0 && threadIdx.x == 0 && (n_elem & 1)) {
        const __hip_bfloat16* b = (const __hip_bfloat16*)in_u32;
        m = fmaxf(m, fabsf(__bfloat162float(b[n_elem - 1])));
    }
    red[threadIdx.x] = m;
    __syncthreads();
    for (int s = 128; s > 0; s >>= 1) {
        if ((int)threadIdx.x < s)
            red[threadIdx.x] = fmaxf(red[threadIdx.x], red[threadIdx.x + s]);
        __syncthreads();
    }
    if (threadIdx.x == 0) atomicMax(slot, __float_as_uint(red[0]));
}

__global__ void fp8lin_quant(const unsigned* __restrict__ in_u32,
                             unsigned long long* __restrict__ out_u64,
                             long n8, const unsigned* __restrict__ hdr, int slot) {
    long i = (long)blockIdx.x * blockDim.x + threadIdx.x;
    if (i >= n8) return;
    float amax = __uint_as_float(hdr[slot]);
    float inv = 1.0f / fmaxf(amax, FP8_EPS);
    const uint4 u = ((const uint4*)in_u32)[i];
    float f[8];
    f[0] = __uint_as_float(u.x << 16) * inv;
    f[1] = __uint_as_float(u.x & 0xFFFF0000u) * inv;
    f[2] = __uint_as_float(u.y << 16) * inv;
    f[3] = __uint_as_float(u.y & 0xFFFF0000u) * inv;
    f[4] = __uint_as_float(u.z << 16) * inv;
    f[5] = __uint_as_float(u.z & 0xFFFF0000u) * inv;
    f[6] = __uint_as_float(u.w << 16) * inv;
    f[7] = __uint_as_float(u.w & 0xFFFF0000u) * inv;
    int w0 = 0, w1 = 0;
    w0 = __builtin_amdgcn_cvt_pk_fp8_f32(f[0], f[1], w0, false);
    w0 = __builtin_amdgcn_cvt_pk_fp8_f32(f[2], f[3], w0, true);
    w1 = __builtin_amdgcn_cvt_pk_fp8_f32(f[4], f[5], w1, false);
    w1 = __builtin_amdgcn_cvt_pk_fp8_f32(f[6], f[7], w1, true);
    out_u64[i] = ((unsigned long long)(unsigned)w1 << 32) | (unsigned)w0;
}

#define BM 128
#define BN 128
#define BK 128
#define PITCH 144

struct AFrag { union { v16i v; unsigned long long u[8]; }; };
struct BFrag { union { v16i v; uint4 q[4]; }; };

__device__ __forceinline__ void stage_tile(const unsigned char* g,
                                           unsigned char* l) {
    unsigned char* gnc = const_cast<unsigned char*>(g);
#pragma unroll
    for (int c = 0; c < 4; ++c) {
#if USE_ASYNC_LDS
        __builtin_amdgcn_global_load_async_to_lds_b128(
            (AS1 v4i_gcc*)(gnc + c * 16), (AS3 v4i_gcc*)(l + c * 16), 0, 0);
#else
        *(uint4*)(l + c * 16) = *(const uint4*)(g + c * 16);
#endif
    }
}

template <int N>
__device__ __forceinline__ void wait_async_le() {
#if USE_ASYNC_LDS
    __builtin_amdgcn_s_wait_asynccnt(N);
#endif
}

__global__ __launch_bounds__(256)
void fp8lin_gemm(const unsigned char* __restrict__ xq,
                 const unsigned char* __restrict__ wq,
                 const __hip_bfloat16* __restrict__ bias,
                 __hip_bfloat16* __restrict__ out,
                 const unsigned* __restrict__ hdr,
                 int M, int N, int K) {
    __shared__ unsigned char ldsA[2][BM * PITCH];
    __shared__ unsigned char ldsB[2][BN * PITCH];

    const int tid  = threadIdx.x;
    const int lane = tid & 31;
    const int wave = tid >> 5;
    const int wm = (wave & 3) * 32;
    const int wn = (wave >> 2) * 64;
    const int m0 = blockIdx.y * BM;
    const int n0 = blockIdx.x * BN;

    const int l16 = lane & 15;
    const int hi  = lane >> 4;
    const int kbA = hi * 16;
    const int khB = hi * 16;

    const int srow = tid >> 1;
    const int scol = (tid & 1) * 64;
    const unsigned char* gA = xq + (size_t)(m0 + srow) * K + scol;
    const unsigned char* gB = wq + (size_t)(n0 + srow) * K + scol;
    unsigned char* lA0 = &ldsA[0][srow * PITCH + scol];
    unsigned char* lB0 = &ldsB[0][srow * PITCH + scol];
    unsigned char* lA1 = &ldsA[1][srow * PITCH + scol];
    unsigned char* lB1 = &ldsB[1][srow * PITCH + scol];

    v8f acc[2][4] = {};

    stage_tile(gA, lA0);
    stage_tile(gB, lB0);

    const int nsteps = K / BK;
    for (int s = 0; s < nsteps; ++s) {
        const int buf = s & 1;
        if (s + 1 < nsteps) {
            const int k = (s + 1) * BK;
            stage_tile(gA + k, buf ? lA0 : lA1);
            stage_tile(gB + k, buf ? lB0 : lB1);
            wait_async_le<8>();
        } else {
            wait_async_le<0>();
        }
        __syncthreads();

        AFrag A[2];
#pragma unroll
        for (int mi = 0; mi < 2; ++mi) {
            const unsigned char* ab = &ldsA[buf][(wm + mi * 16 + l16) * PITCH + kbA];
#pragma unroll
            for (int h = 0; h < 2; ++h)
#pragma unroll
                for (int g = 0; g < 4; ++g)
                    A[mi].u[h * 4 + g] = *(const unsigned long long*)(ab + h * 64 + (g >> 1) * 32 + (g & 1) * 8);
        }
#pragma unroll
        for (int ni = 0; ni < 4; ++ni) {
            BFrag B;
            const unsigned char* bb = &ldsB[buf][(wn + ni * 16 + l16) * PITCH + khB];
#pragma unroll
            for (int g = 0; g < 4; ++g)
                B.q[g] = *(const uint4*)(bb + g * 32);
            acc[0][ni] = __builtin_amdgcn_wmma_f32_16x16x128_fp8_fp8(
                A[0].v, B.v, (short)0, acc[0][ni], false, false);
            acc[1][ni] = __builtin_amdgcn_wmma_f32_16x16x128_fp8_fp8(
                A[1].v, B.v, (short)0, acc[1][ni], false, false);
        }
        __syncthreads();
    }

    const float xs = fmaxf(__uint_as_float(hdr[0]), FP8_EPS);
    const float wsc = fmaxf(__uint_as_float(hdr[1]), FP8_EPS);
    const float sc = xs * wsc;

#pragma unroll
    for (int mi = 0; mi < 2; ++mi) {
#pragma unroll
        for (int ni = 0; ni < 4; ++ni) {
            const int n = n0 + wn + ni * 16 + l16;
            const float bv = __bfloat162float(bias[n]);
#pragma unroll
            for (int v = 0; v < 8; ++v) {
                const int m = m0 + wm + mi * 16 + v + hi * 8;
                float y = acc[mi][ni][v] * sc;
                float yb = __bfloat162float(__float2bfloat16(y));
                out[(size_t)m * N + n] = __float2bfloat16(yb + bv);
            }
        }
    }
}

extern "C" void kernel_launch(void* const* d_in, const int* in_sizes, int n_in,
                              void* d_out, int out_size, void* d_ws, size_t ws_size,
                              hipStream_t stream) {
    const unsigned* x_u32 = (const unsigned*)d_in[0];
    const unsigned* w_u32 = (const unsigned*)d_in[1];
    const __hip_bfloat16* bias = (const __hip_bfloat16*)d_in[2];
    __hip_bfloat16* out = (__hip_bfloat16*)d_out;

    const long nx = in_sizes[0];
    const long nw = in_sizes[1];
    const int  N  = in_sizes[2];
    const int  K  = (int)(nw / N);
    const int  M  = (int)(nx / K);

    unsigned char* ws8 = (unsigned char*)d_ws;
    unsigned* hdr = (unsigned*)ws8;
    unsigned char* xq = ws8 + 256;
    unsigned char* wq = xq + (size_t)M * K;

    fp8lin_init<<<1, 32, 0, stream>>>(hdr);

    fp8lin_amax<<<1024, 256, 0, stream>>>(x_u32, nx, hdr + 0);
    fp8lin_amax<<<256,  256, 0, stream>>>(w_u32, nw, hdr + 1);

    const long nx8 = nx / 8, nw8 = nw / 8;
    fp8lin_quant<<<(unsigned)((nx8 + 255) / 256), 256, 0, stream>>>(
        x_u32, (unsigned long long*)xq, nx8, hdr, 0);
    fp8lin_quant<<<(unsigned)((nw8 + 255) / 256), 256, 0, stream>>>(
        w_u32, (unsigned long long*)wq, nw8, hdr, 1);

    dim3 grid(N / BN, M / BM);
    fp8lin_gemm<<<grid, 256, 0, stream>>>(xq, wq, bias, out, hdr, M, N, K);
}
